// ContextTransferModule_72834055406037
// MI455X (gfx1250) — matrix operand order repaired and checked by reading, not compiled, not run on hardware
//
#include <hip/hip_runtime.h>
#include <cstdint>
#include <cstddef>

#define BB 16
#define CC 32
#define HH 256
#define WW 256
#define HW (HH * WW)
#define KK 128  // 4*C

typedef __attribute__((ext_vector_type(16))) __bf16 v16bf;
typedef __attribute__((ext_vector_type(8)))  float  v8f;

static __device__ __forceinline__ unsigned short f2bf(float f) {
    // round-to-nearest-even f32 -> bf16 bits
    unsigned int u = __float_as_uint(f);
    unsigned int r = u + 0x7FFFu + ((u >> 16) & 1u);
    return (unsigned short)(r >> 16);
}

static __device__ __forceinline__ unsigned int pack2bf(float lo, float hi) {
    return (unsigned int)f2bf(lo) | ((unsigned int)f2bf(hi) << 16);
}

// ---------------------------------------------------------------- sigmoid gates
__global__ void k_sigmoid(const float* __restrict__ w, float* __restrict__ sig, int n) {
    int i = blockIdx.x * blockDim.x + threadIdx.x;
    if (i < n) sig[i] = 1.0f / (1.0f + expf(-w[i]));
}

// ---------------------------------------------------------------- conv_w f32 -> packed bf16 (done once)
__global__ void k_cvt_w(const float* __restrict__ cw, unsigned short* __restrict__ cwbf) {
    int i = blockIdx.x * blockDim.x + threadIdx.x;
    if (i < CC * KK) cwbf[i] = f2bf(cw[i]);
}

// ---------------------------------------------------------------- transpose (B*C planes): [h][w] -> [w][h]
__global__ __launch_bounds__(256) void k_transpose(const float* __restrict__ in,
                                                   float* __restrict__ outT) {
    __shared__ float tile[32][33];
    int bc = blockIdx.z;
    int h0 = blockIdx.y * 32;
    int w0 = blockIdx.x * 32;
    int x = threadIdx.x & 31;
    int y = threadIdx.x >> 5;  // 0..7
    const float* src = in + (size_t)bc * HW;
    float* dst = outT + (size_t)bc * HW;
#pragma unroll
    for (int i = 0; i < 4; ++i) {
        int row = y + i * 8;
        tile[row][x] = src[(size_t)(h0 + row) * WW + (w0 + x)];
    }
    __syncthreads();
#pragma unroll
    for (int i = 0; i < 4; ++i) {
        int row = y + i * 8;
        dst[(size_t)(w0 + row) * HH + (h0 + x)] = tile[x][row];
    }
}

// ================================================================ scans
// Batched LDS staging: R=8 scan steps per flush; write phase packs 4 channels
// into one 8B store (uint2). ctx layout: [pixel p][k], k = dir*32 + c, bf16.
#define RB 8

// ---- H scans (up: dir 2, down: dir 3); block 1024 = (c = t>>5) x (w = t&31)
__global__ __launch_bounds__(1024) void k_scan_h(const float* __restrict__ inp,
                                                 const float* __restrict__ sig,
                                                 unsigned short* __restrict__ ctx) {
    __shared__ float lds[RB][32][33];
    int b  = blockIdx.x >> 3;
    int w0 = (blockIdx.x & 7) * 32;
    int t  = threadIdx.x;
    int wl = t & 31;
    int c  = t >> 5;
    // write-phase mapping
    int c4 = (t & 7) * 4;
    int wp = (t >> 3) & 31;
    int rr = t >> 8;  // 0..3, handles rr and rr+4
    const float* x   = inp + (size_t)(b * CC + c) * HW + (w0 + wl);
    const float* wup = sig;
    const float* wdn = sig + HW;

    // ---- up (forward along h), ctx dir 2
    float state = 0.f;
    for (int hb = 0; hb < HH / RB; ++hb) {
#pragma unroll
        for (int r = 0; r < RB; ++r) {
            int h = hb * RB + r;
            float xv = x[(size_t)h * WW];
            if (h + 1 < HH) __builtin_prefetch(&x[(size_t)(h + 1) * WW], 0, 3);
            state = (h == 0) ? xv : fmaxf(wup[(size_t)(h - 1) * WW + (w0 + wl)] * state + xv, 0.f);
            lds[r][wl][c] = state;
        }
        __syncthreads();
#pragma unroll
        for (int rp = 0; rp < 2; ++rp) {
            int r = rr + rp * 4;
            int h = hb * RB + r;
            size_t p = ((size_t)(b * HH + h)) * WW + (w0 + wp);
            uint2 v;
            v.x = pack2bf(lds[r][wp][c4 + 0], lds[r][wp][c4 + 1]);
            v.y = pack2bf(lds[r][wp][c4 + 2], lds[r][wp][c4 + 3]);
            *(uint2*)(ctx + p * KK + 2 * CC + c4) = v;
        }
        __syncthreads();
    }
    // ---- down (backward along h), ctx dir 3
    state = 0.f;
    for (int hb = HH / RB - 1; hb >= 0; --hb) {
#pragma unroll
        for (int r = RB - 1; r >= 0; --r) {
            int h = hb * RB + r;
            float xv = x[(size_t)h * WW];
            if (h > 0) __builtin_prefetch(&x[(size_t)(h - 1) * WW], 0, 3);
            state = (h == HH - 1) ? xv : fmaxf(wdn[(size_t)(h + 1) * WW + (w0 + wl)] * state + xv, 0.f);
            lds[r][wl][c] = state;
        }
        __syncthreads();
#pragma unroll
        for (int rp = 0; rp < 2; ++rp) {
            int r = rr + rp * 4;
            int h = hb * RB + r;
            size_t p = ((size_t)(b * HH + h)) * WW + (w0 + wp);
            uint2 v;
            v.x = pack2bf(lds[r][wp][c4 + 0], lds[r][wp][c4 + 1]);
            v.y = pack2bf(lds[r][wp][c4 + 2], lds[r][wp][c4 + 3]);
            *(uint2*)(ctx + p * KK + 3 * CC + c4) = v;
        }
        __syncthreads();
    }
}

// ---- W scans on transposed input (left: dir 0, right: dir 1)
// block 1024 = (c = t>>5) x (h = t&31); input inpT layout [b*C + c][w][h]
__global__ __launch_bounds__(1024) void k_scan_w(const float* __restrict__ inpT,
                                                 const float* __restrict__ sig,
                                                 unsigned short* __restrict__ ctx) {
    __shared__ float lds[RB][32][33];
    int b  = blockIdx.x >> 3;
    int h0 = (blockIdx.x & 7) * 32;
    int t  = threadIdx.x;
    int hl = t & 31;
    int c  = t >> 5;
    int c4 = (t & 7) * 4;
    int hp = (t >> 3) & 31;
    int rr = t >> 8;
    const float* x    = inpT + (size_t)(b * CC + c) * HW + (h0 + hl);  // + w*HH
    const float* wlft = sig + 2 * (size_t)HW;
    const float* wrgt = sig + 3 * (size_t)HW;

    // ---- left (forward along w), ctx dir 0
    float state = 0.f;
    for (int wb = 0; wb < WW / RB; ++wb) {
#pragma unroll
        for (int r = 0; r < RB; ++r) {
            int w = wb * RB + r;
            float xv = x[(size_t)w * HH];
            if (w + 1 < WW) __builtin_prefetch(&x[(size_t)(w + 1) * HH], 0, 3);
            state = (w == 0) ? xv : fmaxf(wlft[(size_t)(h0 + hl) * WW + (w - 1)] * state + xv, 0.f);
            lds[r][hl][c] = state;
        }
        __syncthreads();
#pragma unroll
        for (int rp = 0; rp < 2; ++rp) {
            int r = rr + rp * 4;
            int w = wb * RB + r;
            size_t p = ((size_t)(b * HH + h0 + hp)) * WW + w;
            uint2 v;
            v.x = pack2bf(lds[r][hp][c4 + 0], lds[r][hp][c4 + 1]);
            v.y = pack2bf(lds[r][hp][c4 + 2], lds[r][hp][c4 + 3]);
            *(uint2*)(ctx + p * KK + 0 * CC + c4) = v;
        }
        __syncthreads();
    }
    // ---- right (backward along w), ctx dir 1
    state = 0.f;
    for (int wb = WW / RB - 1; wb >= 0; --wb) {
#pragma unroll
        for (int r = RB - 1; r >= 0; --r) {
            int w = wb * RB + r;
            float xv = x[(size_t)w * HH];
            if (w > 0) __builtin_prefetch(&x[(size_t)(w - 1) * HH], 0, 3);
            state = (w == WW - 1) ? xv : fmaxf(wrgt[(size_t)(h0 + hl) * WW + (w + 1)] * state + xv, 0.f);
            lds[r][hl][c] = state;
        }
        __syncthreads();
#pragma unroll
        for (int rp = 0; rp < 2; ++rp) {
            int r = rr + rp * 4;
            int w = wb * RB + r;
            size_t p = ((size_t)(b * HH + h0 + hp)) * WW + w;
            uint2 v;
            v.x = pack2bf(lds[r][hp][c4 + 0], lds[r][hp][c4 + 1]);
            v.y = pack2bf(lds[r][hp][c4 + 2], lds[r][hp][c4 + 3]);
            *(uint2*)(ctx + p * KK + 1 * CC + c4) = v;
        }
        __syncthreads();
    }
}

// ---------------------------------------------------------------- 1x1 conv as WMMA GEMM: M = B*H*W, N = 32, K = 128
// block 256 thr = 8 waves; wave -> 16 pixels x 32 out-channels; block tile = 128 pixels
__global__ __launch_bounds__(256) void k_gemm(const unsigned short* __restrict__ ctx,
                                              const unsigned short* __restrict__ cwbf,  // (32,128) bf16
                                              const float* __restrict__ cb,             // (32)
                                              float* __restrict__ out) {
    __shared__ float lds[CC][129];
    int t    = threadIdx.x;
    int lane = t & 31;
    int wv   = t >> 5;
    int half = lane >> 4;
    int m16  = lane & 15;
    size_t pblock = (size_t)blockIdx.x * 128;
    size_t ptile  = pblock + (size_t)wv * 16;

    // ---- B fragments: pre-packed bf16, lane = N, elem e -> K = half*16 + e (contiguous)
    union BF { v16bf v; uint4 q[2]; };
    BF bfr[2][4];
#pragma unroll
    for (int nt = 0; nt < 2; ++nt) {
        const unsigned short* brow = cwbf + (size_t)(nt * 16 + m16) * KK + half * 16;
#pragma unroll
        for (int kf = 0; kf < 4; ++kf) {
            bfr[nt][kf].q[0] = *(const uint4*)(brow + kf * 32);
            bfr[nt][kf].q[1] = *(const uint4*)(brow + kf * 32 + 8);
        }
    }

    // ---- A fragments from ctx (bf16, row-contiguous K)
    // A layout: lane m16 = M; elem e -> K = half*16 + e (contiguous)  => two 16B runs
    const unsigned short* arow = ctx + (ptile + m16) * KK;
    v8f acc0 = {};
    v8f acc1 = {};
#pragma unroll
    for (int kf = 0; kf < 4; ++kf) {
        BF a;
        a.q[0] = *(const uint4*)(arow + kf * 32 + half * 16);
        a.q[1] = *(const uint4*)(arow + kf * 32 + half * 16 + 8);
        acc0 = __builtin_amdgcn_wmma_f32_16x16x32_bf16(false, a.v, false, bfr[0][kf].v,
                                                       (short)0, acc0, false, false);
        acc1 = __builtin_amdgcn_wmma_f32_16x16x32_bf16(false, a.v, false, bfr[1][kf].v,
                                                       (short)0, acc1, false, false);
    }

    // ---- epilogue: bias + relu, stage in LDS as [c][local_pixel]
    float b0 = cb[m16];
    float b1 = cb[16 + m16];
#pragma unroll
    for (int r = 0; r < 8; ++r) {
        int plocal = wv * 16 + half * 8 + r;  // D layout: M = half*8 + r, N = m16
        lds[m16][plocal]      = fmaxf(acc0[r] + b0, 0.f);
        lds[16 + m16][plocal] = fmaxf(acc1[r] + b1, 0.f);
    }
    __syncthreads();

    // ---- coalesced store: 128 consecutive pixels share (b,h); out is (B,C,H,W)
    int cch = t >> 3;  // 0..31
    int g   = t & 7;   // 16 pixels each
    size_t bidx = pblock / HW;
    size_t s    = pblock - bidx * HW;
    float* dst = out + (bidx * CC + cch) * (size_t)HW + s + g * 16;
#pragma unroll
    for (int j = 0; j < 16; j += 4) {
        float4 v = make_float4(lds[cch][g * 16 + j + 0], lds[cch][g * 16 + j + 1],
                               lds[cch][g * 16 + j + 2], lds[cch][g * 16 + j + 3]);
        *(float4*)(dst + j) = v;
    }
}

// ---------------------------------------------------------------- launcher
extern "C" void kernel_launch(void* const* d_in, const int* in_sizes, int n_in,
                              void* d_out, int out_size, void* d_ws, size_t ws_size,
                              hipStream_t stream) {
    const float* inp = (const float*)d_in[0];   // (B,C,H,W)
    const float* wts = (const float*)d_in[1];   // (4,H,W)
    const float* cw  = (const float*)d_in[2];   // (C,4C)
    const float* cb  = (const float*)d_in[3];   // (C)
    float* out = (float*)d_out;

    char* ws = (char*)d_ws;
    float* sig          = (float*)ws;                                    // 4*HW f32   = 1 MiB
    float* inpT         = (float*)(ws + (size_t)4 * HW * 4);             // B*C*HW f32 = 128 MiB
    unsigned short* ctx = (unsigned short*)(ws + (size_t)4 * HW * 4 +
                                            (size_t)BB * CC * HW * 4);   // B*HW*128 bf16 = 256 MiB
    unsigned short* cwbf = (unsigned short*)(ws + (size_t)4 * HW * 4 +
                                             (size_t)BB * CC * HW * 4 +
                                             (size_t)BB * HW * KK * 2);  // 8 KiB

    k_sigmoid<<<(4 * HW + 1023) / 1024, 1024, 0, stream>>>(wts, sig, 4 * HW);
    k_cvt_w<<<(CC * KK + 255) / 256, 256, 0, stream>>>(cw, cwbf);

    dim3 tg(WW / 32, HH / 32, BB * CC);
    k_transpose<<<tg, 256, 0, stream>>>(inp, inpT);

    k_scan_h<<<BB * (WW / 32), 1024, 0, stream>>>(inp, sig, ctx);
    k_scan_w<<<BB * (HH / 32), 1024, 0, stream>>>(inpT, sig, ctx);

    k_gemm<<<(int)(((size_t)BB * HW) / 128), 256, 0, stream>>>(ctx, cwbf, cb, out);
}
